// GRUCausalMMM_68358699483896
// MI455X (gfx1250) — compile-verified, matrix operand order checked by reading, not run on hardware
//
#include <hip/hip_runtime.h>
#include <math.h>

typedef __attribute__((ext_vector_type(16))) _Float16 v16h;
typedef __attribute__((ext_vector_type(8)))  _Float16 v8h;
typedef __attribute__((ext_vector_type(8)))  float    v8f;

#define BB 256
#define TT 512
#define MM 32
#define CC 16
#define HH 256

__device__ __forceinline__ v8f wmma16(v16h a, v16h b, v8f c) {
  return __builtin_amdgcn_wmma_f32_16x16x32_f16(false, a, false, b, (short)0, c, false, false);
}

__device__ __forceinline__ float sigf(float x) { return 1.0f / (1.0f + __expf(-x)); }
__device__ __forceinline__ float splus(float x) { return (x > 20.f) ? x : log1pf(__expf(x)); }

// A-fragment (16x32 f16, MxK layout): lane holds row=(lane&15); element e<8 -> K=k0+hi*8+e,
// e>=8 -> K=k0+16+hi*8+(e-8).  Source: row-major f32, zero-padded past kmax.
// NOTE: hi must be provably in {0,1} at all call sites so the k<kmax predicate folds.
__device__ __forceinline__ v16h fragA_g(const float* rowp, int k0, int kmax, int hi) {
  v16h r;
#pragma unroll
  for (int e = 0; e < 8; ++e) {
    int k = k0 + hi * 8 + e;
    r[e] = (k < kmax) ? (_Float16)rowp[k] : (_Float16)0.f;
  }
#pragma unroll
  for (int e = 0; e < 8; ++e) {
    int k = k0 + 16 + hi * 8 + e;
    r[e + 8] = (k < kmax) ? (_Float16)rowp[k] : (_Float16)0.f;
  }
  return r;
}

// B-fragment with B[k,n] = W[k*ldn + n] (direct, strided gather), zero-padded past kmax.
__device__ __forceinline__ v16h fragB_dir(const float* W, int ldn, int n, int k0, int kmax, int hi) {
  v16h r;
#pragma unroll
  for (int e = 0; e < 8; ++e) {
    int k = k0 + hi * 8 + e;
    r[e] = (k < kmax) ? (_Float16)W[k * ldn + n] : (_Float16)0.f;
  }
#pragma unroll
  for (int e = 0; e < 8; ++e) {
    int k = k0 + 16 + hi * 8 + e;
    r[e + 8] = (k < kmax) ? (_Float16)W[k * ldn + n] : (_Float16)0.f;
  }
  return r;
}

// Fragment from row-major f16 memory (LDS or global): two contiguous 16B chunks per lane.
__device__ __forceinline__ v16h fragA_h16(const _Float16* rowp, int k0, int hi) {
  v8h x0 = *(const v8h*)(rowp + k0 + hi * 8);
  v8h x1 = *(const v8h*)(rowp + k0 + 16 + hi * 8);
  v16h r;
#pragma unroll
  for (int e = 0; e < 8; ++e) { r[e] = x0[e]; r[e + 8] = x1[e]; }
  return r;
}

// gin fragment pair for K=48 (pad to 64): chunks {hi*8, 16+hi*8, 32+hi*8} valid, last chunk zero.
__device__ __forceinline__ void fragA_gin(const _Float16* rowp, int hi, v16h& a0, v16h& a1) {
  v8h x0 = *(const v8h*)(rowp + hi * 8);
  v8h x1 = *(const v8h*)(rowp + 16 + hi * 8);
  v8h x2 = *(const v8h*)(rowp + 32 + hi * 8);
#pragma unroll
  for (int e = 0; e < 8; ++e) {
    a0[e] = x0[e];
    a0[e + 8] = x1[e];
    a1[e] = x2[e];
    a1[e + 8] = (_Float16)0.f;
  }
}

// ---------------- Kernel 1a: causal encoder -> Z [B*T, 32] f32 ----------------
__global__ __launch_bounds__(32) void enc_kernel(
    const float* __restrict__ Xm, const float* __restrict__ Amat,
    const float* __restrict__ We1, const float* __restrict__ be1,
    const float* __restrict__ We2, const float* __restrict__ be2,
    const float* __restrict__ Wn, const float* __restrict__ bn,
    float* __restrict__ Zb) {
  __shared__ _Float16 sr[16 * 64];   // [send|recv] f16
  __shared__ _Float16 xh[16 * 64];   // [Xm|He] f16
  __shared__ _Float16 e1b[16 * 32];  // relu(SR@We1+be1)
  const int lane = threadIdx.x & 31;
  const int c = lane & 15, hi = (lane >> 4) & 1;
  const size_t row0 = (size_t)blockIdx.x * 16;

  const float* xrow = Xm + (row0 + c) * MM;
  v16h Xa = fragA_g(xrow, 0, MM, hi);
  // Stash Xm (f16) into xh cols 0..31 using the fragment's K pattern.
#pragma unroll
  for (int e = 0; e < 8; ++e) {
    xh[c * 64 + hi * 8 + e] = Xa[e];
    xh[c * 64 + 16 + hi * 8 + e] = Xa[e + 8];
  }
  // send = Xm @ A.T -> B[k,n]=A[n,k]; recv = Xm @ A -> B[k,n]=A[k,n]
#pragma unroll
  for (int nt = 0; nt < 2; ++nt) {
    v16h Bs = fragA_g(Amat + (nt * 16 + c) * MM, 0, MM, hi);
    v8f d = {};
    d = wmma16(Xa, Bs, d);
#pragma unroll
    for (int v = 0; v < 8; ++v) sr[(v + 8 * hi) * 64 + nt * 16 + c] = (_Float16)d[v];
    v16h Br = fragB_dir(Amat, MM, nt * 16 + c, 0, MM, hi);
    v8f d2 = {};
    d2 = wmma16(Xa, Br, d2);
#pragma unroll
    for (int v = 0; v < 8; ++v) sr[(v + 8 * hi) * 64 + 32 + nt * 16 + c] = (_Float16)d2[v];
  }
  __syncthreads();
  // e1 = relu(SR @ We1 + be1)   (K=64)
  v16h As0 = fragA_h16(sr + c * 64, 0, hi);
  v16h As1 = fragA_h16(sr + c * 64, 32, hi);
#pragma unroll
  for (int nt = 0; nt < 2; ++nt) {
    v8f d = {};
    d = wmma16(As0, fragB_dir(We1, MM, nt * 16 + c, 0, 2 * MM, hi), d);
    d = wmma16(As1, fragB_dir(We1, MM, nt * 16 + c, 32, 2 * MM, hi), d);
    float bb = be1[nt * 16 + c];
#pragma unroll
    for (int v = 0; v < 8; ++v) e1b[(v + 8 * hi) * 32 + nt * 16 + c] = (_Float16)fmaxf(d[v] + bb, 0.f);
  }
  __syncthreads();
  // He = relu(e1 @ We2 + be2) -> xh cols 32..63
  v16h Ae = fragA_h16(e1b + c * 32, 0, hi);
#pragma unroll
  for (int nt = 0; nt < 2; ++nt) {
    v8f d = {};
    d = wmma16(Ae, fragB_dir(We2, MM, nt * 16 + c, 0, MM, hi), d);
    float bb = be2[nt * 16 + c];
#pragma unroll
    for (int v = 0; v < 8; ++v) xh[(v + 8 * hi) * 64 + 32 + nt * 16 + c] = (_Float16)fmaxf(d[v] + bb, 0.f);
  }
  __syncthreads();
  // Z = relu([Xm|He] @ Wn + bn) -> global f32
  v16h Ax0 = fragA_h16(xh + c * 64, 0, hi);
  v16h Ax1 = fragA_h16(xh + c * 64, 32, hi);
#pragma unroll
  for (int nt = 0; nt < 2; ++nt) {
    v8f d = {};
    d = wmma16(Ax0, fragB_dir(Wn, MM, nt * 16 + c, 0, 2 * MM, hi), d);
    d = wmma16(Ax1, fragB_dir(Wn, MM, nt * 16 + c, 32, 2 * MM, hi), d);
    float bb = bn[nt * 16 + c];
#pragma unroll
    for (int v = 0; v < 8; ++v)
      Zb[(row0 + v + 8 * hi) * MM + nt * 16 + c] = fmaxf(d[v] + bb, 0.f);
  }
}

// ---------------- Kernel 1b: adstock IIR + normalize + Hill -> hillb f32, gin[:,:,0:32] f16 ----------------
__global__ __launch_bounds__(256) void adstock_kernel(
    const float* __restrict__ Xm, const float* __restrict__ Zb,
    const float* __restrict__ alpha, const float* __restrict__ hill_a,
    const float* __restrict__ hill_g, float* __restrict__ hillb,
    _Float16* __restrict__ gin) {
  int idx = blockIdx.x * blockDim.x + threadIdx.x;
  if (idx >= BB * MM) return;
  int b = idx >> 5, m = idx & 31;
  float a = fminf(fmaxf(alpha[m], 0.f), 1.f);
  float ha = fminf(fmaxf(hill_a[m], 0.1f), 3.f);
  float hg = fminf(fmaxf(hill_g[m], 0.1f), 2.f);
  float hga = __powf(hg, ha);
  const float* x = Xm + (size_t)b * TT * MM + m;
  float* hb = hillb + (size_t)b * TT * MM + m;
  float ad = 0.f, amax = 0.f;
  for (int t = 0; t < TT; ++t) {
    ad = x[(size_t)t * MM] + a * ad;
    hb[(size_t)t * MM] = ad;
    amax = fmaxf(amax, ad);
  }
  float inv = 1.f / fmaxf(amax, 1e-6f);
  const float* z = Zb + (size_t)b * TT * MM + m;
  _Float16* g = gin + (size_t)b * TT * (MM + CC) + m;
  for (int t = 0; t < TT; ++t) {
    float adn = hb[(size_t)t * MM] * inv;
    float num = __powf(fmaxf(adn, 0.f), ha);
    float hill = num / (num + hga + 1e-8f);
    hb[(size_t)t * MM] = hill;
    g[(size_t)t * (MM + CC)] = (_Float16)(hill + z[(size_t)t * MM]);
  }
}

// ---------------- Kernel 1c: ctrl_term + Xc -> gin[:,:,32:48] f16 ----------------
__global__ __launch_bounds__(32) void ctrl_kernel(
    const float* __restrict__ Xc, const float* __restrict__ Wc, const float* __restrict__ bc,
    float* __restrict__ ctrlb, _Float16* __restrict__ gin) {
  const int lane = threadIdx.x & 31;
  const int c = lane & 15, hi = (lane >> 4) & 1;
  const size_t row0 = (size_t)blockIdx.x * 16;
#pragma unroll
  for (int i = 0; i < 8; ++i) {
    int idx = lane * 8 + i;  // 0..255 = 16 rows x 16 ctrl channels
    int r = idx >> 4, cc = idx & 15;
    gin[(row0 + r) * (MM + CC) + MM + cc] = (_Float16)Xc[(row0 + r) * CC + cc];
  }
  const float* xrow = Xc + (row0 + c) * CC;
  v16h Xa = fragA_g(xrow, 0, CC, hi);  // K padded 16->32 with zeros (folds: hi in {0,1})
  float tot[8] = {0, 0, 0, 0, 0, 0, 0, 0};
#pragma unroll
  for (int nt = 0; nt < 16; ++nt) {
    v16h Bc = fragB_dir(Wc, HH, nt * 16 + c, 0, CC, hi);
    v8f d = {};
    d = wmma16(Xa, Bc, d);
    float bb = bc[nt * 16 + c];
#pragma unroll
    for (int v = 0; v < 8; ++v) tot[v] += fmaxf(d[v] + bb, 0.f);
  }
#pragma unroll
  for (int v = 0; v < 8; ++v) {
    float s = tot[v];
    s += __shfl_xor(s, 1); s += __shfl_xor(s, 2); s += __shfl_xor(s, 4); s += __shfl_xor(s, 8);
    if (c == 0) ctrlb[row0 + v + 8 * hi] = 0.3f * s;
  }
}

// ---------------- fused head: w_pos / media_contrib / row-partial y for one step ----------------
__device__ __forceinline__ void head_step(
    int tstep, int nt, const v16h* Ah, const _Float16* wwT,
    int c, int hi, float bwv, const float* __restrict__ hillb,
    float* __restrict__ outW, float* __restrict__ outM,
    float* ypart, float* ypartF, v8f& wp0save, int b0) {
  const int m = nt * 16 + c;
  v8f acc = {};
#pragma unroll
  for (int ks = 0; ks < 8; ++ks) {
    v16h bf = fragA_h16(wwT + m * HH, ks * 32, hi);
    acc = wmma16(Ah[ks], bf, acc);
  }
  v8f wp;
#pragma unroll
  for (int v = 0; v < 8; ++v) wp[v] = splus(acc[v] + bwv);
  if (tstep == 0) wp0save = wp;
  float mc[8];
#pragma unroll
  for (int v = 0; v < 8; ++v) {
    size_t idx = ((size_t)(b0 + v + 8 * hi) * TT + tstep) * MM + m;
    float h = hillb[idx];
    float x = h * wp[v];
    outW[idx] = wp[v];
    outM[idx] = x;
    mc[v] = x;
  }
#pragma unroll
  for (int v = 0; v < 8; ++v) {
    float s = mc[v];
    s += __shfl_xor(s, 1); s += __shfl_xor(s, 2); s += __shfl_xor(s, 4); s += __shfl_xor(s, 8);
    if (c == 0) ypart[nt * 16 + v + 8 * hi] = s;
  }
  if (tstep == 1) {  // fix up t=0: w0 = 0.5*wp1 + 0.5*wp0
    float mc0[8];
#pragma unroll
    for (int v = 0; v < 8; ++v) {
      float w0 = 0.5f * wp[v] + 0.5f * wp0save[v];
      size_t idx0 = ((size_t)(b0 + v + 8 * hi) * TT) * MM + m;
      float h0v = hillb[idx0];
      float x = h0v * w0;
      outW[idx0] = w0;
      outM[idx0] = x;
      mc0[v] = x;
    }
#pragma unroll
    for (int v = 0; v < 8; ++v) {
      float s = mc0[v];
      s += __shfl_xor(s, 1); s += __shfl_xor(s, 2); s += __shfl_xor(s, 4); s += __shfl_xor(s, 8);
      if (c == 0) ypartF[nt * 16 + v + 8 * hi] = s;
    }
  }
}

// ---------------- Kernel 2: GRU scan + fused head.  16 WGs x 16 waves.
// Wave w owns hidden columns [w*16, w*16+16) for all three gates; Wh/Wi B-fragments
// live in registers for the entire T loop; only the 8KB f16 h tile round-trips LDS.
__global__ __launch_bounds__(512) void gru_kernel(
    const _Float16* __restrict__ gin, const float* __restrict__ hillb,
    const float* __restrict__ ctrlb,
    const float* __restrict__ Wi, const float* __restrict__ bi,
    const float* __restrict__ Wh, const float* __restrict__ bh,
    const float* __restrict__ Ww, const float* __restrict__ bw,
    const float* __restrict__ reg_emb, const float* __restrict__ bias,
    const float* __restrict__ h0,
    float* __restrict__ outY, float* __restrict__ outW, float* __restrict__ outM) {
  __shared__ _Float16 hbuf[16 * HH];   // h_{t-1} f16, row-major [16][256]
  __shared__ _Float16 wwT[MM * HH];    // Ww transposed: wwT[m][k] = Ww[k][m]
  __shared__ float ypart[2 * 16];
  __shared__ float ypartF[2 * 16];
  __shared__ float s_misc;
  const int tid = threadIdx.x;
  const int wave = (tid >> 5) & 15;
  const int lane = tid & 31;
  const int c = lane & 15;
  const int hi = (lane >> 4) & 1;
  const int b0 = blockIdx.x * 16;

  for (int i = tid; i < 16 * HH; i += 512) hbuf[i] = (_Float16)h0[i & (HH - 1)];
  for (int i = tid; i < MM * HH; i += 512) {
    int m = i >> 8, k = i & 255;
    wwT[i] = (_Float16)Ww[k * MM + m];
  }
  if (tid == 0) {
    float s = 0.f;
    for (int k = 0; k < HH; ++k) s += reg_emb[k];  // row 0 of reg_emb
    s_misc = 0.3f * s + bias[0];
  }
  const int j = wave * 16 + c;  // hidden column owned by this lane
  // Register-resident B fragments: gh[.,j] = sum_k h[.,k]*Wh[gate*H+j, k]
  v16h BhR[8], BhZ[8], BhN[8];
#pragma unroll
  for (int ks = 0; ks < 8; ++ks) {
    BhR[ks] = fragA_g(Wh + (size_t)(j) * HH, ks * 32, HH, hi);
    BhZ[ks] = fragA_g(Wh + (size_t)(HH + j) * HH, ks * 32, HH, hi);
    BhN[ks] = fragA_g(Wh + (size_t)(2 * HH + j) * HH, ks * 32, HH, hi);
  }
  v16h BiR[2], BiZ[2], BiN[2];
#pragma unroll
  for (int ks = 0; ks < 2; ++ks) {
    BiR[ks] = fragA_g(Wi + (size_t)(j) * (MM + CC), ks * 32, MM + CC, hi);
    BiZ[ks] = fragA_g(Wi + (size_t)(HH + j) * (MM + CC), ks * 32, MM + CC, hi);
    BiN[ks] = fragA_g(Wi + (size_t)(2 * HH + j) * (MM + CC), ks * 32, MM + CC, hi);
  }
  const float biR = bi[j], biZ = bi[HH + j], biN = bi[2 * HH + j];
  const float bhR = bh[j], bhZ = bh[HH + j], bhN = bh[2 * HH + j];
  const float bwv = (wave < 2) ? bw[wave * 16 + c] : 0.f;
  const float h0j = h0[j];
  v8f hreg;
#pragma unroll
  for (int v = 0; v < 8; ++v) hreg[v] = h0j;
  v8f wp0save = {};
  __syncthreads();

  for (int t = 0; t < TT; ++t) {
    v16h Ah[8];
#pragma unroll
    for (int ks = 0; ks < 8; ++ks) Ah[ks] = fragA_h16(hbuf + c * HH, ks * 32, hi);
    const _Float16* grow = gin + ((size_t)(b0 + c) * TT + t) * (MM + CC);
    v16h Ag0, Ag1;
    fragA_gin(grow, hi, Ag0, Ag1);
    // head for step t-1 (uses Ah == h_{t-1}); wave-uniform branch keeps EXEC full
    if (wave < 2 && t >= 1)
      head_step(t - 1, wave, Ah, wwT, c, hi, bwv, hillb, outW, outM, ypart, ypartF, wp0save, b0);
    v8f ar = {}, az = {}, ain = {}, ahn = {};
    ar = wmma16(Ag0, BiR[0], ar); ar = wmma16(Ag1, BiR[1], ar);
    az = wmma16(Ag0, BiZ[0], az); az = wmma16(Ag1, BiZ[1], az);
    ain = wmma16(Ag0, BiN[0], ain); ain = wmma16(Ag1, BiN[1], ain);
#pragma unroll
    for (int ks = 0; ks < 8; ++ks) {
      ar = wmma16(Ah[ks], BhR[ks], ar);
      az = wmma16(Ah[ks], BhZ[ks], az);
      ahn = wmma16(Ah[ks], BhN[ks], ahn);
    }
    v8f hnew;
#pragma unroll
    for (int v = 0; v < 8; ++v) {
      float r = sigf(ar[v] + biR + bhR);
      float z = sigf(az[v] + biZ + bhZ);
      float n = tanhf(ain[v] + biN + r * (ahn[v] + bhN));
      hnew[v] = (1.f - z) * n + z * hreg[v];
    }
    hreg = hnew;
    __syncthreads();  // all waves done reading hbuf; ypart for t-1 valid
    if (t >= 1 && tid < 16) {
      size_t rb = (size_t)(b0 + tid) * TT;
      outY[rb + (t - 1)] = ypart[tid] + ypart[16 + tid] + ctrlb[rb + (t - 1)] + s_misc;
      if (t == 2)  // rewrite y[b,0] with fixed w0
        outY[rb] = ypartF[tid] + ypartF[16 + tid] + ctrlb[rb] + s_misc;
    }
#pragma unroll
    for (int v = 0; v < 8; ++v) hbuf[(v + 8 * hi) * HH + j] = (_Float16)hnew[v];
    __syncthreads();  // hbuf now holds h_t
  }
  {  // epilogue: head for t = T-1
    v16h Ah[8];
#pragma unroll
    for (int ks = 0; ks < 8; ++ks) Ah[ks] = fragA_h16(hbuf + c * HH, ks * 32, hi);
    if (wave < 2)
      head_step(TT - 1, wave, Ah, wwT, c, hi, bwv, hillb, outW, outM, ypart, ypartF, wp0save, b0);
    __syncthreads();
    if (tid < 16) {
      size_t rb = (size_t)(b0 + tid) * TT;
      outY[rb + (TT - 1)] = ypart[tid] + ypart[16 + tid] + ctrlb[rb + (TT - 1)] + s_misc;
    }
  }
}

extern "C" void kernel_launch(void* const* d_in, const int* in_sizes, int n_in,
                              void* d_out, int out_size, void* d_ws, size_t ws_size,
                              hipStream_t stream) {
  const float* Xm = (const float*)d_in[0];
  const float* Xc = (const float*)d_in[1];
  // d_in[2] = R (unused: reference indexes zeros, not R)
  const float* Amat = (const float*)d_in[3];
  const float* We1 = (const float*)d_in[4];
  const float* be1 = (const float*)d_in[5];
  const float* We2 = (const float*)d_in[6];
  const float* be2 = (const float*)d_in[7];
  const float* Wn = (const float*)d_in[8];
  const float* bn = (const float*)d_in[9];
  const float* Wi = (const float*)d_in[10];
  const float* bi = (const float*)d_in[11];
  const float* Wh = (const float*)d_in[12];
  const float* bh = (const float*)d_in[13];
  const float* Ww = (const float*)d_in[14];
  const float* bw = (const float*)d_in[15];
  const float* alpha = (const float*)d_in[16];
  const float* hill_a = (const float*)d_in[17];
  const float* hill_g = (const float*)d_in[18];
  const float* Wc = (const float*)d_in[19];
  const float* bc = (const float*)d_in[20];
  const float* reg_emb = (const float*)d_in[21];
  const float* bias = (const float*)d_in[22];
  const float* h0 = (const float*)d_in[23];

  // workspace layout (all 16B aligned): Z[B*T*32]f32 | hill[B*T*32]f32 | gin[B*T*48]f16 | ctrl[B*T]f32
  float* Zb = (float*)d_ws;
  float* hillb = Zb + (size_t)BB * TT * MM;
  _Float16* gin = (_Float16*)(hillb + (size_t)BB * TT * MM);
  float* ctrlb = (float*)(gin + (size_t)BB * TT * (MM + CC));

  float* outY = (float*)d_out;
  float* outW = outY + (size_t)BB * TT;
  float* outM = outW + (size_t)BB * TT * MM;

  enc_kernel<<<(BB * TT) / 16, 32, 0, stream>>>(Xm, Amat, We1, be1, We2, be2, Wn, bn, Zb);
  ctrl_kernel<<<(BB * TT) / 16, 32, 0, stream>>>(Xc, Wc, bc, ctrlb, gin);
  adstock_kernel<<<(BB * MM) / 256, 256, 0, stream>>>(Xm, Zb, alpha, hill_a, hill_g, hillb, gin);
  gru_kernel<<<BB / 16, 512, 0, stream>>>(gin, hillb, ctrlb, Wi, bi, Wh, bh, Ww, bw,
                                          reg_emb, bias, h0, outY, outW, outM);
}
